// EncoderBlock_4466765988170
// MI455X (gfx1250) — matrix operand order repaired and checked by reading, not compiled, not run on hardware
//
#include <hip/hip_runtime.h>

typedef __attribute__((ext_vector_type(16))) __bf16 v16bf;
typedef __attribute__((ext_vector_type(8)))  float  v8f;

union FragBF {
  v16bf v;
  unsigned int u[8];
};

__device__ __forceinline__ unsigned short f32_to_bf16(float f) {
  unsigned int u = __float_as_uint(f);
  u += 0x7FFFu + ((u >> 16) & 1u);   // round-to-nearest-even
  return (unsigned short)(u >> 16);
}

// ---------------------------------------------------------------------------
// RMSNorm: y = g * x / (||x||/sqrt(d) + eps); writes f32 (residual) + bf16 (GEMM A)
// One block per row of 1024, 256 threads, 4 elems/thread.
// ---------------------------------------------------------------------------
__global__ __launch_bounds__(256)
void rmsnorm_kernel(const float* __restrict__ x, const float* __restrict__ g,
                    float* __restrict__ y32, unsigned short* __restrict__ y16) {
  const int C = 1024;
  const int row = blockIdx.x;
  const float* xr = x + (size_t)row * C;
  float vals[4];
  float ss = 0.f;
#pragma unroll
  for (int i = 0; i < 4; ++i) {
    vals[i] = xr[threadIdx.x + i * 256];
    ss += vals[i] * vals[i];
  }
#pragma unroll
  for (int m = 16; m >= 1; m >>= 1) ss += __shfl_xor(ss, m, 32);
  __shared__ float red[8];
  const int wave = threadIdx.x >> 5;
  const int lane = threadIdx.x & 31;
  if (lane == 0) red[wave] = ss;
  __syncthreads();
  float tot = red[0];
#pragma unroll
  for (int w = 1; w < 8; ++w) tot += red[w];
  const float rms = sqrtf(tot) * 0.03125f;  // * 1/sqrt(1024)
  const float inv = 1.0f / (rms + 1e-8f);
#pragma unroll
  for (int i = 0; i < 4; ++i) {
    const int col = threadIdx.x + i * 256;
    const float yv = g[col] * vals[i] * inv;
    y32[(size_t)row * C + col] = yv;
    y16[(size_t)row * C + col] = f32_to_bf16(yv);
  }
}

// ---------------------------------------------------------------------------
// Tiled bf16 WMMA GEMM, double-buffered LDS:
//   out = epilogue(A_bf16[M,K] @ W_f32[K,N])
// 128x128 block tile, 8 waves, each wave 32x64 (2x4 of 16x16x32 WMMA).
// A tile (128x32 bf16) is fetched with CDNA5 async global->LDS B64 DMA
// (ASYNCcnt) into the ping-pong buffer while the current tile computes.
// W tile (32x128 f32) is register-staged (needs f32->bf16 conversion) and
// stored transposed after the current tile's WMMAs issue.
// As row stride = 36 halfwords (72 B) so every async B64 lands 8B-aligned.
// ---------------------------------------------------------------------------
template <bool HAS_BIAS, bool GELU, bool RES, bool OUT_BF16>
__global__ __launch_bounds__(256)
void gemm_bf16_kernel(const unsigned short* __restrict__ A,  // [M,K] bf16
                      const float* __restrict__ W,           // [K,N] f32
                      const float* __restrict__ bias,        // [N]  f32
                      const float* __restrict__ res,         // [M,N] f32
                      void* __restrict__ outp,               // [M,N]
                      int M, int N, int K) {
  __shared__ unsigned short As[2][128 * 36];  // [m][k], stride 36 (8B-aligned rows)
  __shared__ unsigned short Bs[2][128 * 34];  // transposed: [n][k], stride 34
  const int tid = threadIdx.x;
  const int wave = tid >> 5, lane = tid & 31;
  const int lm = lane & 15, hi = lane >> 4;
  const int wm = wave & 3, wn = wave >> 2;
  const int bm = blockIdx.y * 128, bn = blockIdx.x * 128;
  (void)M;

  const v8f vzero = {0.f, 0.f, 0.f, 0.f, 0.f, 0.f, 0.f, 0.f};
  v8f c[2][4];
#pragma unroll
  for (int mi = 0; mi < 2; ++mi)
#pragma unroll
    for (int ni = 0; ni < 4; ++ni) c[mi][ni] = vzero;

  // Async A-tile stage: 128 rows x 32 bf16 = 1024 x B64, 4 per thread.
  // GVS addressing: saddr = uniform tile base, vaddr = per-lane byte offset.
  auto stageA = [&](int k0, int bufi) {
    const unsigned short* sbase = A + (size_t)bm * K + k0;
#pragma unroll
    for (int i = 0; i < 4; ++i) {
      const int p = tid + i * 256;
      const int row = p >> 3, ch = p & 7;
      const unsigned lds =
          (unsigned)(size_t)(const void*)&As[bufi][row * 36 + ch * 4];
      const int voff = (row * K + ch * 4) * 2;  // bytes
      asm volatile("global_load_async_to_lds_b64 %0, %1, %2"
                   :: "v"(lds), "v"(voff), "s"(sbase)
                   : "memory");
    }
  };

  float wreg[16];
  auto loadW = [&](int k0) {
#pragma unroll
    for (int i = 0; i < 16; ++i) {
      const int p = tid + i * 256;
      wreg[i] = W[(size_t)(k0 + (p >> 7)) * N + bn + (p & 127)];
    }
  };
  auto storeW = [&](int bufi) {
#pragma unroll
    for (int i = 0; i < 16; ++i) {
      const int p = tid + i * 256;
      Bs[bufi][(p & 127) * 34 + (p >> 7)] = f32_to_bf16(wreg[i]);
    }
  };

  // prologue: stage tile 0
  stageA(0, 0);
  loadW(0);
  storeW(0);
  asm volatile("s_wait_asynccnt 0" ::: "memory");
  __syncthreads();

  const int ktiles = K >> 5;
  int buf = 0;
  for (int kt = 0; kt < ktiles; ++kt) {
    const bool more = (kt + 1) < ktiles;
    if (more) {
      stageA((kt + 1) << 5, buf ^ 1);  // async DMA overlaps with WMMA below
      loadW((kt + 1) << 5);
    }

    const unsigned int* AsU = (const unsigned int*)&As[buf][0];
    const unsigned int* BsU = (const unsigned int*)&Bs[buf][0];
    FragBF a[2], b[4];
#pragma unroll
    for (int mi = 0; mi < 2; ++mi) {
      const int row = wm * 32 + mi * 16 + lm;
#pragma unroll
      for (int v = 0; v < 8; ++v) {
        const int kk = ((v & 3) << 1) + (hi << 3) + ((v >> 2) << 4);
        a[mi].u[v] = AsU[row * 18 + (kk >> 1)];
      }
    }
#pragma unroll
    for (int ni = 0; ni < 4; ++ni) {
      const int col = wn * 64 + ni * 16 + lm;
#pragma unroll
      for (int v = 0; v < 8; ++v) {
        const int kk = ((v & 3) << 1) + (hi << 3) + ((v >> 2) << 4);
        b[ni].u[v] = BsU[col * 17 + (kk >> 1)];
      }
    }
#pragma unroll
    for (int mi = 0; mi < 2; ++mi)
#pragma unroll
      for (int ni = 0; ni < 4; ++ni)
        c[mi][ni] = __builtin_amdgcn_wmma_f32_16x16x32_bf16(
            false, a[mi].v, false, b[ni].v, (short)0, c[mi][ni], false, false);

    if (more) {
      storeW(buf ^ 1);                              // fill next W buffer
      asm volatile("s_wait_asynccnt 0" ::: "memory");  // next A tile landed
    }
    __syncthreads();  // everyone done reading buf / writing buf^1
    buf ^= 1;
  }

#pragma unroll
  for (int mi = 0; mi < 2; ++mi)
#pragma unroll
    for (int ni = 0; ni < 4; ++ni)
#pragma unroll
      for (int r = 0; r < 8; ++r) {
        const int row = bm + wm * 32 + mi * 16 + (hi << 3) + r;
        const int col = bn + wn * 64 + ni * 16 + lm;
        float v = c[mi][ni][r];
        if constexpr (HAS_BIAS) v += bias[col];
        if constexpr (GELU) v = 0.5f * v * (1.0f + erff(v * 0.70710678118654752f));
        if constexpr (RES) v += res[(size_t)row * N + col];
        if constexpr (OUT_BF16)
          ((unsigned short*)outp)[(size_t)row * N + col] = f32_to_bf16(v);
        else
          ((float*)outp)[(size_t)row * N + col] = v;
      }
}

// ---------------------------------------------------------------------------
// Flash attention: grid = (T/128, B*H). 8 waves/block, 16 queries per wave.
// Streams 64-key tiles through LDS; S=QK^T and O+=PV via bf16 WMMA; online
// softmax stats in f32 with half-wave shfl reductions (rows stay within a
// 16-lane half on wave32).
// ---------------------------------------------------------------------------
__global__ __launch_bounds__(256)
void attn_kernel(const unsigned short* __restrict__ qkv,  // [B,T,3*1024] bf16
                 unsigned short* __restrict__ ctx) {      // [B,T,1024] bf16
  const int T = 2048;
  __shared__ unsigned short Kt[64 * 66];     // [key][d]
  __shared__ unsigned short Vt[64 * 66];     // [d][key] (transposed)
  __shared__ unsigned short Pw[8][16 * 66];  // per-wave P: [m][key]
  const int tid = threadIdx.x;
  const int wave = tid >> 5, lane = tid & 31;
  const int lm = lane & 15, hi = lane >> 4;
  const int bb = blockIdx.y >> 4, hh = blockIdx.y & 15;
  const int q0 = blockIdx.x * 128 + wave * 16;
  const unsigned int* qkvU = (const unsigned int*)qkv;
  unsigned int* KtU = (unsigned int*)Kt;
  const unsigned int* VtU = (const unsigned int*)Vt;
  const unsigned int* PwU = (const unsigned int*)(&Pw[wave][0]);

  FragBF qf[2];  // Q 16x64 = two 16x32 A-fragments
#pragma unroll
  for (int f = 0; f < 2; ++f)
#pragma unroll
    for (int v = 0; v < 8; ++v) {
      const int k = ((v & 3) << 1) + (hi << 3) + ((v >> 2) << 4) + (f << 5);
      const size_t addr = (size_t)(bb * T + q0 + lm) * 3072 + hh * 64 + k;
      qf[f].u[v] = qkvU[addr >> 1];
    }

  const v8f vzero = {0.f, 0.f, 0.f, 0.f, 0.f, 0.f, 0.f, 0.f};
  v8f o[4];
#pragma unroll
  for (int ni = 0; ni < 4; ++ni) o[ni] = vzero;
  float rowM[8], rowL[8];
#pragma unroll
  for (int r = 0; r < 8; ++r) { rowM[r] = -3.0e38f; rowL[r] = 0.f; }

  for (int ktile = 0; ktile < T / 64; ++ktile) {
    const int kb = ktile * 64;
    __syncthreads();
#pragma unroll
    for (int i = 0; i < 8; ++i) {  // stage K (row-major) and V (transposed)
      const int p = tid + i * 256;
      const int key = p >> 5, dc = p & 31;
      const size_t base = (size_t)(bb * T + kb + key) * 3072 + hh * 64;
      KtU[key * 33 + dc] = qkvU[(base + 1024 + dc * 2) >> 1];
      const unsigned int pv = qkvU[(base + 2048 + dc * 2) >> 1];
      Vt[(2 * dc) * 66 + key] = (unsigned short)(pv & 0xFFFFu);
      Vt[(2 * dc + 1) * 66 + key] = (unsigned short)(pv >> 16);
    }
    __syncthreads();

    v8f s[4];
#pragma unroll
    for (int ni = 0; ni < 4; ++ni) s[ni] = vzero;
#pragma unroll
    for (int ni = 0; ni < 4; ++ni)
#pragma unroll
      for (int f = 0; f < 2; ++f) {  // S = Q @ K^T (B[k=d][n=key] = Kt[key][d])
        FragBF kf;
#pragma unroll
        for (int v = 0; v < 8; ++v) {
          const int d = ((v & 3) << 1) + (hi << 3) + ((v >> 2) << 4) + (f << 5);
          kf.u[v] = KtU[(ni * 16 + lm) * 33 + (d >> 1)];
        }
        s[ni] = __builtin_amdgcn_wmma_f32_16x16x32_bf16(
            false, qf[f].v, false, kf.v, (short)0, s[ni], false, false);
      }

#pragma unroll
    for (int r = 0; r < 8; ++r) {  // online softmax per query row
      float xv[4];
      float mx = -3.0e38f;
#pragma unroll
      for (int ni = 0; ni < 4; ++ni) {
        xv[ni] = s[ni][r] * 0.125f;  // HEAD_DIM^-0.5
        mx = fmaxf(mx, xv[ni]);
      }
#pragma unroll
      for (int msk = 8; msk >= 1; msk >>= 1) mx = fmaxf(mx, __shfl_xor(mx, msk, 32));
      const float nm = fmaxf(rowM[r], mx);
      const float alpha = __expf(rowM[r] - nm);
      float sum = 0.f;
#pragma unroll
      for (int ni = 0; ni < 4; ++ni) {
        const float pv = __expf(xv[ni] - nm);
        xv[ni] = pv;
        sum += pv;
      }
#pragma unroll
      for (int msk = 8; msk >= 1; msk >>= 1) sum += __shfl_xor(sum, msk, 32);
      rowL[r] = rowL[r] * alpha + sum;
      rowM[r] = nm;
#pragma unroll
      for (int ni = 0; ni < 4; ++ni) o[ni][r] = o[ni][r] * alpha;
      const int m = (hi << 3) + r;
#pragma unroll
      for (int ni = 0; ni < 4; ++ni)
        Pw[wave][m * 66 + ni * 16 + lm] = f32_to_bf16(xv[ni]);
    }
    // wave-local LDS write->cross-lane-read ordering (DS ops are in-order,
    // the wait + memory clobber stops compiler reordering)
    asm volatile("s_wait_dscnt 0" ::: "memory");

    FragBF pa[2];  // P as A-fragments (k = key)
#pragma unroll
    for (int f = 0; f < 2; ++f)
#pragma unroll
      for (int v = 0; v < 8; ++v) {
        const int kkey = ((v & 3) << 1) + (hi << 3) + ((v >> 2) << 4) + (f << 5);
        pa[f].u[v] = PwU[lm * 33 + (kkey >> 1)];
      }
#pragma unroll
    for (int ni = 0; ni < 4; ++ni)
#pragma unroll
      for (int f = 0; f < 2; ++f) {  // O += P @ V (B[k=key][n=d] = Vt[d][key])
        FragBF vf;
#pragma unroll
        for (int v = 0; v < 8; ++v) {
          const int kkey = ((v & 3) << 1) + (hi << 3) + ((v >> 2) << 4) + (f << 5);
          vf.u[v] = VtU[(ni * 16 + lm) * 33 + (kkey >> 1)];
        }
        o[ni] = __builtin_amdgcn_wmma_f32_16x16x32_bf16(
            false, pa[f].v, false, vf.v, (short)0, o[ni], false, false);
      }
  }

#pragma unroll
  for (int ni = 0; ni < 4; ++ni)
#pragma unroll
    for (int r = 0; r < 8; ++r) {
      const int t = q0 + (hi << 3) + r;
      const int d = ni * 16 + lm;
      const float v = o[ni][r] / rowL[r];
      ctx[(size_t)(bb * T + t) * 1024 + hh * 64 + d] = f32_to_bf16(v);
    }
}

// ---------------------------------------------------------------------------
// Pipeline: rmsnorm1 -> QKV gemm -> attention -> out-proj(+bias+res) ->
//           rmsnorm2 (h3 f32 into d_out) -> ffn1(+bias+gelu) -> ffn2(+bias+res)
// Workspace layout (peak 56 MB) with stream-ordered reuse:
//   [0,16M)  h1_f32      (live ph1-4)    reused by hidden_bf16 (ph6-7)
//   [16,24M) h1_bf16     (live ph1-2)    reused by hidden_bf16
//   [24,48M) qkv_bf16    (live ph2-3)    reused by h2_f32 [24,40M) ph4-5
//                                        and   h3_bf16 [40,48M) ph5-6
//   [48,56M) ctx_bf16    (live ph3-4)
// ---------------------------------------------------------------------------
extern "C" void kernel_launch(void* const* d_in, const int* in_sizes, int n_in,
                              void* d_out, int out_size, void* d_ws, size_t ws_size,
                              hipStream_t stream) {
  const float* x     = (const float*)d_in[0];
  const float* w_in  = (const float*)d_in[1];
  const float* w_out = (const float*)d_in[2];
  const float* b_out = (const float*)d_in[3];
  const float* w1    = (const float*)d_in[4];
  const float* b1    = (const float*)d_in[5];
  const float* w2    = (const float*)d_in[6];
  const float* b2    = (const float*)d_in[7];
  const float* g1    = (const float*)d_in[8];
  const float* g2    = (const float*)d_in[9];
  float* out = (float*)d_out;

  char* ws = (char*)d_ws;
  const size_t MB = 1024u * 1024u;
  float*          h1f  = (float*)(ws + 0 * MB);
  unsigned short* h1b  = (unsigned short*)(ws + 16 * MB);
  unsigned short* qkvb = (unsigned short*)(ws + 24 * MB);
  unsigned short* ctxb = (unsigned short*)(ws + 48 * MB);
  float*          h2f  = (float*)(ws + 24 * MB);
  unsigned short* h3b  = (unsigned short*)(ws + 40 * MB);
  unsigned short* hidb = (unsigned short*)(ws + 0 * MB);

  const dim3 blk(256);
  rmsnorm_kernel<<<4096, blk, 0, stream>>>(x, g1, h1f, h1b);
  gemm_bf16_kernel<false, false, false, true><<<dim3(24, 32), blk, 0, stream>>>(
      h1b, w_in, nullptr, nullptr, (void*)qkvb, 4096, 3072, 1024);
  attn_kernel<<<dim3(16, 32), blk, 0, stream>>>(qkvb, ctxb);
  gemm_bf16_kernel<true, false, true, false><<<dim3(8, 32), blk, 0, stream>>>(
      ctxb, w_out, b_out, h1f, (void*)h2f, 4096, 1024, 1024);
  rmsnorm_kernel<<<4096, blk, 0, stream>>>(h2f, g2, out, h3b);
  gemm_bf16_kernel<true, true, false, true><<<dim3(32, 32), blk, 0, stream>>>(
      h3b, w1, b1, nullptr, (void*)hidb, 4096, 4096, 1024);
  gemm_bf16_kernel<true, false, true, false><<<dim3(8, 32), blk, 0, stream>>>(
      hidb, w2, b2, out, (void*)out, 4096, 1024, 4096);
  (void)in_sizes; (void)n_in; (void)out_size; (void)ws_size;
}
